// SetKernelMultiheadAttention_55310588838251
// MI455X (gfx1250) — compile-verified, matrix operand order checked by reading, not run on hardware
//
#include <hip/hip_runtime.h>
#include <hip/hip_bf16.h>

#define E_DIM   1024
#define H_NUM   16
#define DH_DIM  64
#define TK_NUM  16
#define RANK_N  32
#define GAMMA_C 0.3f
#define RIDGE_C 0.0001f
#define B_NUM   2
#define L_NUM   2048
#define BH_NUM  (B_NUM * H_NUM)

#define CDIV(a, b) (((a) + (b) - 1) / (b))

typedef __attribute__((ext_vector_type(16))) __bf16 v16bf;
typedef __attribute__((ext_vector_type(8)))  float  v8f;

union FragU { v16bf v; uint4 u[2]; };

// ---- CDNA5 async global->LDS path (ASYNCcnt-tracked), with safe fallback ----
#if defined(__has_builtin)
#  if __has_builtin(__builtin_amdgcn_global_load_async_to_lds_b128) && \
      __has_builtin(__builtin_amdgcn_s_wait_asynccnt)
#    define USE_ASYNC_LDS 1
#  endif
#endif

// Builtin signature (from clang diagnostic): first param is
// 'int __attribute__((vector_size(16))) __device__ *' (AS1), second is the
// LDS-side pointer (AS3), then imm offset and imm cpol.
typedef int v4i_t __attribute__((vector_size(16)));
typedef v4i_t as1_v4i __attribute__((address_space(1)));
typedef v4i_t as3_v4i __attribute__((address_space(3)));

__device__ inline void stage16(const unsigned short* gsrc, unsigned short* ldst) {
#if defined(USE_ASYNC_LDS)
  __builtin_amdgcn_global_load_async_to_lds_b128((as1_v4i*)gsrc, (as3_v4i*)ldst,
                                                 0, 0);
#else
  *(uint4*)ldst = *(const uint4*)gsrc;
#endif
}

__device__ inline void stage_wait() {
#if defined(USE_ASYNC_LDS)
  __builtin_amdgcn_s_wait_asynccnt(0);
#endif
}

__device__ inline unsigned short f2bf(float x) {
  unsigned int u = __builtin_bit_cast(unsigned int, x);
  unsigned int r = u + 0x7FFFu + ((u >> 16) & 1u);  // round-to-nearest-even
  return (unsigned short)(r >> 16);
}

// Load a 16x32 bf16 fragment (A layout, or mirrored B layout) from a row-major
// tile. Lane l<16 -> row l, K in {0..7, 16..23}; lane l>=16 -> row l-16,
// K in {8..15, 24..31}. A and B receive the same K permutation, which cancels
// in the contraction.
__device__ inline v16bf load_frag(const unsigned short* base, int stride) {
  int l  = threadIdx.x & 31;
  int r  = l & 15;
  int ko = (l >> 4) * 8;
  const unsigned short* p = base + r * stride + ko;
  FragU f;
  f.u[0] = *(const uint4*)(p);       // K = ko .. ko+7
  f.u[1] = *(const uint4*)(p + 16);  // K = 16+ko .. 16+ko+7
  return f.v;
}

__device__ inline float hmax16(float x) {
  #pragma unroll
  for (int m = 1; m < 16; m <<= 1) x = fmaxf(x, __shfl_xor(x, m, 16));
  return x;
}
__device__ inline float hadd16(float x) {
  #pragma unroll
  for (int m = 1; m < 16; m <<= 1) x += __shfl_xor(x, m, 16);
  return x;
}

#define WMMA_BF16(A, B, C) \
  __builtin_amdgcn_wmma_f32_16x16x32_bf16(false, (A), false, (B), (short)0, (C), false, false)

// ---------------------------------------------------------------------------
// f32 -> bf16 convert
// ---------------------------------------------------------------------------
__global__ void f32_to_bf16_kernel(const float* __restrict__ src,
                                   unsigned short* __restrict__ dst, int n) {
  int i = blockIdx.x * 256 + threadIdx.x;
  if (i < n) dst[i] = f2bf(src[i]);
}

// ---------------------------------------------------------------------------
// C[M,N] = A[M,K](bf16) @ W[N,K]^T(bf16) + bias[N], fp32 out.
// Block 256 thr = 8 waves (4 M x 2 N), tile 128 x 128, K-step 32.
// Wave tile 32 x 64 -> 8 WMMAs per K chunk from 6 fragment loads.
// A/W tiles staged via async global->LDS (ASYNCcnt) when available.
// ---------------------------------------------------------------------------
__global__ __launch_bounds__(256) void gemm_bias_kernel(
    const unsigned short* __restrict__ A, const unsigned short* __restrict__ W,
    const float* __restrict__ bias, float* __restrict__ C,
    int M, int N, int K) {
  __shared__ __align__(16) unsigned short As[128 * 40];
  __shared__ __align__(16) unsigned short Ws[128 * 40];

  int tid = threadIdx.x;
  int wid = tid >> 5;
  int bm = blockIdx.x * 128;
  int bn = blockIdx.y * 128;
  int wm = (wid & 3) * 32;   // 4 waves along M
  int wn = (wid >> 2) * 64;  // 2 waves along N

  v8f c[2][4];
  #pragma unroll
  for (int i = 0; i < 2; i++)
    #pragma unroll
    for (int j = 0; j < 4; j++) c[i][j] = (v8f){};

  // Each thread owns two fixed 16B chunks of each 128x32 tile (512 chunks).
  int r0 = tid >> 2, c0 = (tid & 3) * 8;
  int r1 = (tid + 256) >> 2, c1 = ((tid + 256) & 3) * 8;

  for (int k0 = 0; k0 < K; k0 += 32) {
    stage16(&A[(size_t)(bm + r0) * K + k0 + c0], &As[r0 * 40 + c0]);
    stage16(&A[(size_t)(bm + r1) * K + k0 + c1], &As[r1 * 40 + c1]);
    stage16(&W[(size_t)(bn + r0) * K + k0 + c0], &Ws[r0 * 40 + c0]);
    stage16(&W[(size_t)(bn + r1) * K + k0 + c1], &Ws[r1 * 40 + c1]);
    if (k0 + 32 < K) {  // gfx1250 global_prefetch_b8 for the next K tile
      __builtin_prefetch(&A[(size_t)(bm + r0) * K + k0 + 32 + c0], 0, 3);
      __builtin_prefetch(&W[(size_t)(bn + r0) * K + k0 + 32 + c0], 0, 3);
    }
    stage_wait();
    __syncthreads();

    v16bf a0 = load_frag(&As[(wm + 0) * 40], 40);
    v16bf a1 = load_frag(&As[(wm + 16) * 40], 40);
    v16bf b0 = load_frag(&Ws[(wn + 0) * 40], 40);
    v16bf b1 = load_frag(&Ws[(wn + 16) * 40], 40);
    v16bf b2 = load_frag(&Ws[(wn + 32) * 40], 40);
    v16bf b3 = load_frag(&Ws[(wn + 48) * 40], 40);

    c[0][0] = WMMA_BF16(a0, b0, c[0][0]);
    c[0][1] = WMMA_BF16(a0, b1, c[0][1]);
    c[0][2] = WMMA_BF16(a0, b2, c[0][2]);
    c[0][3] = WMMA_BF16(a0, b3, c[0][3]);
    c[1][0] = WMMA_BF16(a1, b0, c[1][0]);
    c[1][1] = WMMA_BF16(a1, b1, c[1][1]);
    c[1][2] = WMMA_BF16(a1, b2, c[1][2]);
    c[1][3] = WMMA_BF16(a1, b3, c[1][3]);
    __syncthreads();
  }

  // C fragment layout: lane l, elem i -> row (l>=16 ? 8 : 0)+i, col l&15.
  int l = tid & 31;
  int n = l & 15;
  int rb = (l >> 4) * 8;
  int row0 = bm + wm, col0 = bn + wn;
  #pragma unroll
  for (int j = 0; j < 4; j++) {
    float bv = bias[col0 + j * 16 + n];
    #pragma unroll
    for (int i = 0; i < 8; i++) {
      C[(size_t)(row0 + rb + i) * N + col0 + j * 16 + n]      = c[0][j][i] + bv;
      C[(size_t)(row0 + 16 + rb + i) * N + col0 + j * 16 + n] = c[1][j][i] + bv;
    }
  }
}

// ---------------------------------------------------------------------------
// Top-16-of-64 -> 64-bit set mask per (b,h,l). One thread per row.
// ---------------------------------------------------------------------------
__global__ void topk_mask_kernel(const float* __restrict__ x,
                                 unsigned long long* __restrict__ mask) {
  int idx = blockIdx.x * 256 + threadIdx.x;
  if (idx >= B_NUM * L_NUM * H_NUM) return;
  int h = idx % H_NUM;
  int bl = idx / H_NUM;
  int b = bl / L_NUM, lpos = bl % L_NUM;
  const float* row = x + (size_t)bl * E_DIM + h * DH_DIM;

  float v[DH_DIM];
  #pragma unroll
  for (int d = 0; d < DH_DIM; d++) v[d] = row[d];

  unsigned long long used = 0ull;
  for (int t = 0; t < TK_NUM; t++) {
    float best = -3.4e38f;
    int bi = 0;
    for (int d = 0; d < DH_DIM; d++) {
      bool free_slot = ((used >> d) & 1ull) == 0ull;
      if (free_slot && v[d] > best) { best = v[d]; bi = d; }
    }
    used |= 1ull << bi;
  }
  mask[((size_t)(b * H_NUM + h)) * L_NUM + lpos] = used;
}

// ---------------------------------------------------------------------------
// Per (b,h): build k_zz (32x32) from landmark masks, Jacobi eigh in LDS,
// e = evecs / sqrt(max(evals, ridge)). One wave per matrix, lane-per-column.
// ---------------------------------------------------------------------------
__global__ __launch_bounds__(32) void kzz_eigh_kernel(
    const unsigned long long* __restrict__ kmask, float* __restrict__ e_out) {
  __shared__ float Ash[32 * 33];
  __shared__ float Vsh[32 * 33];
  __shared__ unsigned long long lm[RANK_N];

  int bh = blockIdx.x;
  int t = threadIdx.x;

  int land = (int)((float)t * (float)(L_NUM - 1) / (float)(RANK_N - 1) + 0.5f);
  lm[t] = kmask[(size_t)bh * L_NUM + land];
  __syncthreads();

  for (int j = 0; j < RANK_N; j++) {
    int inter = __popcll(lm[t] & lm[j]);
    float delta = fmaxf(2.0f * TK_NUM - 2.0f * (float)inter, 0.0f);
    Ash[t * 33 + j] = __expf(-GAMMA_C * delta) + (t == j ? RIDGE_C : 0.0f);
    Vsh[t * 33 + j] = (t == j) ? 1.0f : 0.0f;
  }
  __syncthreads();

  for (int sweep = 0; sweep < 12; sweep++) {
    for (int p = 0; p < RANK_N - 1; p++) {
      for (int q = p + 1; q < RANK_N; q++) {
        float apq = Ash[p * 33 + q];
        float app = Ash[p * 33 + p];
        float aqq = Ash[q * 33 + q];
        __syncthreads();
        float c = 1.0f, s = 0.0f;
        if (fabsf(apq) > 1e-12f) {
          float tau = (aqq - app) / (2.0f * apq);
          float tt = (tau >= 0.0f ? 1.0f : -1.0f) /
                     (fabsf(tau) + sqrtf(1.0f + tau * tau));
          c = rsqrtf(1.0f + tt * tt);
          s = tt * c;
        }
        float rp = Ash[p * 33 + t], rq = Ash[q * 33 + t];
        Ash[p * 33 + t] = c * rp - s * rq;
        Ash[q * 33 + t] = s * rp + c * rq;
        __syncthreads();
        float cp = Ash[t * 33 + p], cq = Ash[t * 33 + q];
        Ash[t * 33 + p] = c * cp - s * cq;
        Ash[t * 33 + q] = s * cp + c * cq;
        float vp = Vsh[t * 33 + p], vq = Vsh[t * 33 + q];
        Vsh[t * 33 + p] = c * vp - s * vq;
        Vsh[t * 33 + q] = s * vp + c * vq;
        __syncthreads();
      }
    }
  }

  float ev = Ash[t * 33 + t];
  float inv = rsqrtf(fmaxf(ev, RIDGE_C));
  for (int i = 0; i < RANK_N; i++)
    e_out[(size_t)bh * 1024 + i * RANK_N + t] = Vsh[i * 33 + t] * inv;
}

// ---------------------------------------------------------------------------
// phi[l][r] = sum_j LUT[popc(mask[l] & land[j])] * e[j][r]; bf16 out [BH][L][32]
// ---------------------------------------------------------------------------
__global__ __launch_bounds__(256) void phi_kernel(
    const unsigned long long* __restrict__ mask,
    const unsigned long long* __restrict__ land_src,
    const float* __restrict__ emat, unsigned short* __restrict__ phi) {
  __shared__ float eL[1024];
  __shared__ unsigned long long lm[RANK_N];
  __shared__ float lut[TK_NUM + 1];

  int bh = blockIdx.x;
  int t = threadIdx.x;
  if (t < RANK_N) {
    int land = (int)((float)t * (float)(L_NUM - 1) / (float)(RANK_N - 1) + 0.5f);
    lm[t] = land_src[(size_t)bh * L_NUM + land];
  }
  if (t < TK_NUM + 1)
    lut[t] = __expf(-GAMMA_C * (2.0f * TK_NUM - 2.0f * (float)t));
  for (int i = t; i < 1024; i += 256) eL[i] = emat[(size_t)bh * 1024 + i];
  __syncthreads();

  int lidx = blockIdx.y * 256 + t;
  unsigned long long qm = mask[(size_t)bh * L_NUM + lidx];
  float kr[RANK_N];
  #pragma unroll
  for (int j = 0; j < RANK_N; j++) kr[j] = lut[__popcll(qm & lm[j])];

  unsigned short* out = phi + ((size_t)bh * L_NUM + lidx) * RANK_N;
  for (int r = 0; r < RANK_N; r++) {
    float s = 0.0f;
    #pragma unroll
    for (int j = 0; j < RANK_N; j++) s += kr[j] * eL[j * RANK_N + r];
    out[r] = f2bf(s);
  }
}

// ---------------------------------------------------------------------------
// vT[b,h,d,l] (bf16) = v_f32[b,l,h*64+d] : makes V columns contiguous for
// B-fragment loads in the attention kernel.
// ---------------------------------------------------------------------------
__global__ void v_transpose_kernel(const float* __restrict__ v,
                                   unsigned short* __restrict__ vT) {
  int i = blockIdx.x * 256 + threadIdx.x;
  if (i >= B_NUM * H_NUM * DH_DIM * L_NUM) return;
  int lpos = i % L_NUM;
  int r = i / L_NUM;           // bh*64 + d
  int d = r % DH_DIM;
  int bh = r / DH_DIM;
  int b = bh / H_NUM, h = bh % H_NUM;
  vT[i] = f2bf(v[((size_t)b * L_NUM + lpos) * E_DIM + h * DH_DIM + d]);
}

// ---------------------------------------------------------------------------
// Flash attention: per wave a 16-row q tile; stream k in chunks of 32.
// scores via 2 WMMAs (K=RANK=32), online softmax via 16-lane shuffles,
// P staged through LDS into an A fragment, out += P @ V via 4 WMMAs (DH=64).
// ---------------------------------------------------------------------------
__global__ __launch_bounds__(128) void attention_kernel(
    const unsigned short* __restrict__ phiq,
    const unsigned short* __restrict__ phik,
    const unsigned short* __restrict__ vT, float* __restrict__ attn) {
  __shared__ __align__(16) unsigned short p_lds[4 * 16 * 40];

  int bh = blockIdx.x;
  int qt = blockIdx.y;
  int tid = threadIdx.x;
  int wid = tid >> 5;
  int l = tid & 31;
  int b = bh / H_NUM, h = bh % H_NUM;

  const unsigned short* pq  = phiq + (size_t)bh * L_NUM * RANK_N;
  const unsigned short* pk  = phik + (size_t)bh * L_NUM * RANK_N;
  const unsigned short* vtb = vT + (size_t)bh * DH_DIM * L_NUM;

  int qrow0 = qt * 64 + wid * 16;
  v16bf aq = load_frag(pq + (size_t)qrow0 * RANK_N, RANK_N);

  v8f m, ls, acc0 = {}, acc1 = {}, acc2 = {}, acc3 = {};
  #pragma unroll
  for (int i = 0; i < 8; i++) { m[i] = -3.0e38f; ls[i] = 0.0f; }

  unsigned short* pw = &p_lds[wid * 16 * 40];
  int n = l & 15, rb = (l >> 4) * 8;
  v8f zero = {};

  for (int kc = 0; kc < L_NUM; kc += 32) {
    v16bf bk0 = load_frag(pk + (size_t)kc * RANK_N, RANK_N);
    v16bf bk1 = load_frag(pk + (size_t)(kc + 16) * RANK_N, RANK_N);
    v8f s0 = WMMA_BF16(aq, bk0, zero);
    v8f s1 = WMMA_BF16(aq, bk1, zero);

    v8f p0, p1;
    #pragma unroll
    for (int i = 0; i < 8; i++) {
      float t = fmaxf(s0[i], s1[i]);
      t = hmax16(t);
      float mn = fmaxf(m[i], t);
      float sc = __expf(m[i] - mn);
      p0[i] = __expf(s0[i] - mn);
      p1[i] = __expf(s1[i] - mn);
      float rs = hadd16(p0[i] + p1[i]);
      ls[i] = ls[i] * sc + rs;
      acc0[i] *= sc; acc1[i] *= sc; acc2[i] *= sc; acc3[i] *= sc;
      m[i] = mn;
      pw[(rb + i) * 40 + n]      = f2bf(p0[i]);
      pw[(rb + i) * 40 + 16 + n] = f2bf(p1[i]);
    }
    __syncthreads();

    v16bf pa = load_frag(pw, 40);
    v16bf bv0 = load_frag(vtb + (size_t)0 * 16 * L_NUM + kc, L_NUM);
    v16bf bv1 = load_frag(vtb + (size_t)1 * 16 * L_NUM + kc, L_NUM);
    v16bf bv2 = load_frag(vtb + (size_t)2 * 16 * L_NUM + kc, L_NUM);
    v16bf bv3 = load_frag(vtb + (size_t)3 * 16 * L_NUM + kc, L_NUM);
    acc0 = WMMA_BF16(pa, bv0, acc0);
    acc1 = WMMA_BF16(pa, bv1, acc1);
    acc2 = WMMA_BF16(pa, bv2, acc2);
    acc3 = WMMA_BF16(pa, bv3, acc3);
    __syncthreads();
  }

  #pragma unroll
  for (int i = 0; i < 8; i++) {
    float inv = 1.0f / ls[i];
    int grow = qrow0 + rb + i;
    size_t base = ((size_t)b * L_NUM + grow) * E_DIM + h * DH_DIM + n;
    attn[base + 0]  = acc0[i] * inv;
    attn[base + 16] = acc1[i] * inv;
    attn[base + 32] = acc2[i] * inv;
    attn[base + 48] = acc3[i] * inv;
  }
}

// ---------------------------------------------------------------------------
extern "C" void kernel_launch(void* const* d_in, const int* in_sizes, int n_in,
                              void* d_out, int out_size, void* d_ws, size_t ws_size,
                              hipStream_t stream) {
  (void)in_sizes; (void)n_in; (void)out_size; (void)ws_size;
  const float* query = (const float*)d_in[0];
  const float* key   = (const float*)d_in[1];
  const float* value = (const float*)d_in[2];
  const float* Wq = (const float*)d_in[3];
  const float* bq = (const float*)d_in[4];
  const float* Wk = (const float*)d_in[5];
  const float* bk = (const float*)d_in[6];
  const float* Wv = (const float*)d_in[7];
  const float* bv = (const float*)d_in[8];
  const float* Wo = (const float*)d_in[9];
  const float* bo = (const float*)d_in[10];

  const size_t NX = (size_t)B_NUM * L_NUM * E_DIM;  // 4.19M
  const size_t NW = (size_t)E_DIM * E_DIM;          // 1.05M
  const int M = B_NUM * L_NUM;                      // 4096

  char* ws = (char*)d_ws;
  size_t off = 0;
  auto alloc = [&](size_t bytes) -> void* {
    void* p = ws + off;
    off = (off + bytes + 255) & ~(size_t)255;
    return p;
  };

  unsigned short* wq_bf = (unsigned short*)alloc(NW * 2);
  unsigned short* wk_bf = (unsigned short*)alloc(NW * 2);
  unsigned short* wv_bf = (unsigned short*)alloc(NW * 2);
  unsigned short* wo_bf = (unsigned short*)alloc(NW * 2);
  unsigned short* xq_bf = (unsigned short*)alloc(NX * 2);
  unsigned short* xk_bf = (unsigned short*)alloc(NX * 2);
  unsigned short* xv_bf = (unsigned short*)alloc(NX * 2);
  float* qf = (float*)alloc(NX * 4);
  float* kf = (float*)alloc(NX * 4);
  float* vf = (float*)alloc(NX * 4);
  unsigned long long* qmask = (unsigned long long*)alloc((size_t)BH_NUM * L_NUM * 8);
  unsigned long long* kmask = (unsigned long long*)alloc((size_t)BH_NUM * L_NUM * 8);
  float* emat = (float*)alloc((size_t)BH_NUM * 1024 * 4);
  unsigned short* phiq = (unsigned short*)alloc((size_t)BH_NUM * L_NUM * RANK_N * 2);
  unsigned short* phik = (unsigned short*)alloc((size_t)BH_NUM * L_NUM * RANK_N * 2);
  unsigned short* vT = (unsigned short*)alloc((size_t)BH_NUM * DH_DIM * L_NUM * 2);
  float* attnf = (float*)alloc(NX * 4);
  unsigned short* attn_bf = (unsigned short*)alloc(NX * 2);

  // 1) Convert activations and weights to bf16.
  f32_to_bf16_kernel<<<CDIV((int)NX, 256), 256, 0, stream>>>(query, xq_bf, (int)NX);
  f32_to_bf16_kernel<<<CDIV((int)NX, 256), 256, 0, stream>>>(key,   xk_bf, (int)NX);
  f32_to_bf16_kernel<<<CDIV((int)NX, 256), 256, 0, stream>>>(value, xv_bf, (int)NX);
  f32_to_bf16_kernel<<<CDIV((int)NW, 256), 256, 0, stream>>>(Wq, wq_bf, (int)NW);
  f32_to_bf16_kernel<<<CDIV((int)NW, 256), 256, 0, stream>>>(Wk, wk_bf, (int)NW);
  f32_to_bf16_kernel<<<CDIV((int)NW, 256), 256, 0, stream>>>(Wv, wv_bf, (int)NW);
  f32_to_bf16_kernel<<<CDIV((int)NW, 256), 256, 0, stream>>>(Wo, wo_bf, (int)NW);

  // 2) Q/K/V projections (WMMA GEMMs), tile 128x128.
  dim3 gg(M / 128, E_DIM / 128);
  gemm_bias_kernel<<<gg, 256, 0, stream>>>(xq_bf, wq_bf, bq, qf, M, E_DIM, E_DIM);
  gemm_bias_kernel<<<gg, 256, 0, stream>>>(xk_bf, wk_bf, bk, kf, M, E_DIM, E_DIM);
  gemm_bias_kernel<<<gg, 256, 0, stream>>>(xv_bf, wv_bf, bv, vf, M, E_DIM, E_DIM);

  // 3) Top-k set masks.
  int nrows = B_NUM * L_NUM * H_NUM;
  topk_mask_kernel<<<CDIV(nrows, 256), 256, 0, stream>>>(qf, qmask);
  topk_mask_kernel<<<CDIV(nrows, 256), 256, 0, stream>>>(kf, kmask);

  // 4) k_zz + Jacobi eigh + whitening factor e.
  kzz_eigh_kernel<<<BH_NUM, 32, 0, stream>>>(kmask, emat);

  // 5) phi_q, phi_k (bf16).
  dim3 pg(BH_NUM, L_NUM / 256);
  phi_kernel<<<pg, 256, 0, stream>>>(qmask, kmask, emat, phiq);
  phi_kernel<<<pg, 256, 0, stream>>>(kmask, kmask, emat, phik);

  // 6) V transpose to [BH, DH, L] bf16.
  int nvt = B_NUM * H_NUM * DH_DIM * L_NUM;
  v_transpose_kernel<<<CDIV(nvt, 256), 256, 0, stream>>>(vf, vT);

  // 7) Flash attention (WMMA scores + WMMA P@V).
  dim3 ag(BH_NUM, L_NUM / 64);
  attention_kernel<<<ag, 128, 0, stream>>>(phiq, phik, vT, attnf);

  // 8) Output projection into d_out.
  f32_to_bf16_kernel<<<CDIV((int)NX, 256), 256, 0, stream>>>(attnf, attn_bf, (int)NX);
  gemm_bias_kernel<<<gg, 256, 0, stream>>>(attn_bf, wo_bf, bo, (float*)d_out,
                                           M, E_DIM, E_DIM);
}
